// MoELayerLoadBalance_1468878815865
// MI455X (gfx1250) — matrix operand order repaired and checked by reading, not compiled, not run on hardware
//
#include <hip/hip_runtime.h>

// ---------------------------------------------------------------------------
// MoE layer (top-2 of 8 experts), D=1024, FF=4096, T=8192 tokens, fp32 I/O.
// Memory-bound (268MB fp32 weights): read fp32 weights once, convert to bf16
// in registers, accumulate fp32 via v_wmma_f32_16x16x32_bf16 (wave32).
// 512-thread blocks (16 waves), M=32 token tile, each wave owns a 64-col
// output slice -> 64 accumulator VGPRs/lane; all 16 B-fragment loads stay in
// flight (single L2 round-trip per fragment). Combine weights hoisted to
// registers per expert.
// ---------------------------------------------------------------------------

typedef __attribute__((ext_vector_type(16))) __bf16 v16bf;
typedef __attribute__((ext_vector_type(8)))  __bf16 v8bf;
typedef __attribute__((ext_vector_type(8)))  float  v8f;

#define D_MODEL  1024
#define FF_DIM   4096
#define NEXP     8
#define TOKENS   8192
#define TILE_M   32         // tokens per block (2 x 16-row WMMA sub-tiles)
#define NTHREADS 512        // 16 waves
#define FF_CHUNK 256        // hidden chunk = 16 waves x 16 cols

// -------------------------- gating kernel ----------------------------------
__global__ __launch_bounds__(256) void moe_gate_kernel(
    const float* __restrict__ x, const float* __restrict__ gate_w,
    const float* __restrict__ gate_b, const float* __restrict__ expert_bias,
    float* __restrict__ combine)
{
  const int lane = threadIdx.x & 31;
  const int wv   = threadIdx.x >> 5;
  const int t    = blockIdx.x * 8 + wv;

  float acc[NEXP];
#pragma unroll
  for (int e = 0; e < NEXP; ++e) acc[e] = 0.f;

  const float* xr = x + (size_t)t * D_MODEL;
  for (int d = lane; d < D_MODEL; d += 32) {
    const float xv = xr[d];
    const float* gw = gate_w + d * NEXP;
#pragma unroll
    for (int e = 0; e < NEXP; ++e) acc[e] += xv * gw[e];
  }
#pragma unroll
  for (int off = 16; off >= 1; off >>= 1) {
#pragma unroll
    for (int e = 0; e < NEXP; ++e) acc[e] += __shfl_xor(acc[e], off, 32);
  }

  if (lane == 0) {
    float probs[NEXP], logits[NEXP];
#pragma unroll
    for (int e = 0; e < NEXP; ++e) {
      const float s = acc[e] + gate_b[e];
      probs[e]  = 1.f / (1.f + __expf(-s));  // sigmoid of raw score
      logits[e] = s + expert_bias[e];        // bias affects routing only
    }
    int i0 = 0;
#pragma unroll
    for (int e = 1; e < NEXP; ++e) if (logits[e] > logits[i0]) i0 = e;
    int i1 = (i0 == 0) ? 1 : 0;
#pragma unroll
    for (int e = 0; e < NEXP; ++e)
      if (e != i0 && logits[e] > logits[i1]) i1 = e;
    const float w0 = probs[i0], w1 = probs[i1];
    const float inv = 1.f / (w0 + w1);
    float* cr = combine + (size_t)t * NEXP;
#pragma unroll
    for (int e = 0; e < NEXP; ++e) cr[e] = 0.f;
    cr[i0] = w0 * inv;
    cr[i1] = w1 * inv;
  }
}

// 16-bit A-fragment (16x32, ISA 7.12.2): per lane the 16 elements are two
// contiguous 8-element runs -> two 16B LDS reads.
//   j=0..7  -> K = 8*hf + j
//   j=8..15 -> K = 16 + 8*hf + (j-8)
__device__ __forceinline__ v16bf load_afrag(const __bf16* base) {
  const v8bf lo = *(const v8bf*)(base);
  const v8bf hi = *(const v8bf*)(base + 16);
  return __builtin_shufflevector(lo, hi, 0, 1, 2, 3, 4, 5, 6, 7,
                                 8, 9, 10, 11, 12, 13, 14, 15);
}

// -------------------------- fused expert FFN -------------------------------
// Grid: T/32 blocks of 512 threads (16 waves). Block owns a 32-token tile,
// loops experts (uniform skip on all-zero tile combine), fuses
// h = relu(x@w1+b1) and out += combine .* (h@w2), with the combine row-scale
// folded into h (A rows live in one lane). b2 term added per expert.
__global__ __launch_bounds__(NTHREADS) void moe_ffn_kernel(
    const float* __restrict__ x,  const float* __restrict__ w1,
    const float* __restrict__ b1, const float* __restrict__ w2,
    const float* __restrict__ b2, const float* __restrict__ combine,
    float* __restrict__ out)
{
  __shared__ __bf16 x_lds[TILE_M * D_MODEL];   // 64 KB: x tile in bf16
  __shared__ __bf16 h_lds[TILE_M * FF_CHUNK];  // 16 KB: scaled hidden chunk
  __shared__ float  comb_lds[TILE_M * NEXP];   //  1 KB

  const int tid  = threadIdx.x;
  const int lane = tid & 31;
  const int wv   = tid >> 5;        // wave 0..15 -> owns D cols [wv*64, +64)
  const int hf   = lane >> 4;       // lane half (0/1)
  const int ln   = lane & 15;
  const int tok0 = blockIdx.x * TILE_M;

  // ---- stage x tile into LDS as bf16 (float4 global reads) ----
  for (int i = tid; i < TILE_M * (D_MODEL / 4); i += NTHREADS) {
    const int row = i >> 8;                       // D_MODEL/4 == 256 per row
    const int c4  = (i & 255) << 2;
    const float4 v = *(const float4*)(x + (size_t)(tok0 + row) * D_MODEL + c4);
    __bf16* dst = &x_lds[row * D_MODEL + c4];
    dst[0] = (__bf16)v.x; dst[1] = (__bf16)v.y;
    dst[2] = (__bf16)v.z; dst[3] = (__bf16)v.w;
  }
  if (tid < TILE_M * NEXP) comb_lds[tid] = combine[(size_t)tok0 * NEXP + tid];
  __syncthreads();

  // persistent output accumulators: 32 x 1024 across block, 8 v8f per lane
  v8f acc0[4], acc1[4];
  const v8f vzero = {};
#pragma unroll
  for (int f = 0; f < 4; ++f) { acc0[f] = vzero; acc1[f] = vzero; }

  for (int e = 0; e < NEXP; ++e) {
    // uniform expert skip (routing): all threads compute identical sum
    float csum = 0.f;
#pragma unroll
    for (int m = 0; m < TILE_M; ++m) csum += __builtin_fabsf(comb_lds[m * NEXP + e]);
    if (csum == 0.f) continue;

    // hoist this lane's combine row-weights for expert e (C rows: r + 8*hf)
    float cw0[8], cw1[8];
#pragma unroll
    for (int r = 0; r < 8; ++r) {
      const int m = r + (hf << 3);
      cw0[r] = comb_lds[m * NEXP + e];
      cw1[r] = comb_lds[(16 + m) * NEXP + e];
    }

    const float* w1e = w1 + (size_t)e * D_MODEL * FF_DIM;
    const float* w2e = w2 + (size_t)e * FF_DIM * D_MODEL;

    for (int ff = 0; ff < FF_DIM; ff += FF_CHUNK) {
      // ---- stage 1: wave computes h tiles [2 x 16x16] at cols ff+wv*16 ----
      v8f hc0 = vzero, hc1 = vzero;
      const int col1 = ff + wv * 16 + ln;
      for (int kk = 0; kk < D_MODEL; kk += 32) {
        const v16bf a0 = load_afrag(&x_lds[ln * D_MODEL + kk + hf * 8]);
        const v16bf a1 = load_afrag(&x_lds[(16 + ln) * D_MODEL + kk + hf * 8]);
        v16bf b;
#pragma unroll
        for (int j = 0; j < 16; ++j)           // B: K = 8*hf + (j&7) + 16*(j>>3), N = ln
          b[j] = (__bf16)w1e[(size_t)(kk + hf * 8 + (j & 7) + ((j >> 3) << 4)) * FF_DIM + col1];
        hc0 = __builtin_amdgcn_wmma_f32_16x16x32_bf16(
            false, a0, false, b, (short)0, hc0, false, false);
        hc1 = __builtin_amdgcn_wmma_f32_16x16x32_bf16(
            false, a1, false, b, (short)0, hc1, false, false);
      }
      // bias + relu + fold combine row-scale, park chunk in LDS as bf16
      {
        const float b1v = b1[e * FF_DIM + col1];
#pragma unroll
        for (int r = 0; r < 8; ++r) {          // C: M = r + 8*hf, N = ln
          const int m = r + (hf << 3);
          float h0 = hc0[r] + b1v;
          float h1 = hc1[r] + b1v;
          h0 = h0 > 0.f ? h0 : 0.f;
          h1 = h1 > 0.f ? h1 : 0.f;
          h0 *= cw0[r];
          h1 *= cw1[r];
          h_lds[m * FF_CHUNK + wv * 16 + ln]        = (__bf16)h0;
          h_lds[(16 + m) * FF_CHUNK + wv * 16 + ln] = (__bf16)h1;
        }
      }
      __syncthreads();

      // ---- stage 2: acc += h[32 x FF_CHUNK] @ w2[ff:+FF_CHUNK, wv*64:+64] --
      for (int k2 = 0; k2 < FF_CHUNK; k2 += 32) {
        const v16bf a0 = load_afrag(&h_lds[ln * FF_CHUNK + k2 + hf * 8]);
        const v16bf a1 = load_afrag(&h_lds[(16 + ln) * FF_CHUNK + k2 + hf * 8]);
#pragma unroll
        for (int f = 0; f < 4; ++f) {
          const int nb = wv * 64 + f * 16 + ln;
          v16bf b;
#pragma unroll
          for (int j = 0; j < 16; ++j)
            b[j] = (__bf16)w2e[(size_t)(ff + k2 + hf * 8 + (j & 7) + ((j >> 3) << 4)) * D_MODEL + nb];
          acc0[f] = __builtin_amdgcn_wmma_f32_16x16x32_bf16(
              false, a0, false, b, (short)0, acc0[f], false, false);
          acc1[f] = __builtin_amdgcn_wmma_f32_16x16x32_bf16(
              false, a1, false, b, (short)0, acc1[f], false, false);
        }
      }
      __syncthreads();  // protect h_lds before next chunk overwrites it
    }

    // + combine .* b2[e] (row-scaled bias of second linear)
#pragma unroll
    for (int f = 0; f < 4; ++f) {
      const float bv = b2[e * D_MODEL + wv * 64 + f * 16 + ln];
#pragma unroll
      for (int r = 0; r < 8; ++r) {
        acc0[f][r] += cw0[r] * bv;
        acc1[f][r] += cw1[r] * bv;
      }
    }
  }

  // ---- store the 32 x 1024 tile (each element owned by exactly one lane) ----
#pragma unroll
  for (int f = 0; f < 4; ++f) {
    const int n = wv * 64 + f * 16 + ln;
#pragma unroll
    for (int r = 0; r < 8; ++r) {
      const int m = r + (hf << 3);
      out[(size_t)(tok0 + m) * D_MODEL + n]      = acc0[f][r];
      out[(size_t)(tok0 + 16 + m) * D_MODEL + n] = acc1[f][r];
    }
  }
}

// ---------------------------------------------------------------------------
extern "C" void kernel_launch(void* const* d_in, const int* in_sizes, int n_in,
                              void* d_out, int out_size, void* d_ws, size_t ws_size,
                              hipStream_t stream) {
  const float* x  = (const float*)d_in[0];
  const float* w1 = (const float*)d_in[1];
  const float* b1 = (const float*)d_in[2];
  const float* w2 = (const float*)d_in[3];
  const float* b2 = (const float*)d_in[4];
  const float* gw = (const float*)d_in[5];
  const float* gb = (const float*)d_in[6];
  const float* eb = (const float*)d_in[7];
  float* out = (float*)d_out;
  float* combine = (float*)d_ws;  // TOKENS * NEXP floats = 256 KB scratch

  moe_gate_kernel<<<TOKENS / 8, 256, 0, stream>>>(x, gw, gb, eb, combine);
  moe_ffn_kernel<<<TOKENS / TILE_M, NTHREADS, 0, stream>>>(x, w1, b1, w2, b2, combine, out);
}
